// PLE_65781719106223
// MI455X (gfx1250) — matrix operand order repaired and checked by reading, not compiled, not run on hardware
//
#include <hip/hip_runtime.h>

// ---------------- types ----------------
typedef __attribute__((ext_vector_type(16))) __bf16 v16bf;
typedef __attribute__((ext_vector_type(2)))  __bf16 v2bf;
typedef __attribute__((ext_vector_type(8)))  float  v8f;
typedef __attribute__((ext_vector_type(8)))  unsigned short u16x8;
typedef __attribute__((ext_vector_type(4)))  float f32x4;

union frag16 {
  struct { u16x8 lo, hi; } p;
  v16bf h;
};

__device__ __forceinline__ unsigned short f2bf(float f) {
  unsigned int u = __float_as_uint(f);
  u += 0x7FFFu + ((u >> 16) & 1u);          // round-to-nearest-even
  return (unsigned short)(u >> 16);
}
// packed 2x f32 -> 2x bf16 in one u32 (low = a, high = b)
__device__ __forceinline__ unsigned int f2bf_pk(float a, float b) {
#if __has_builtin(__builtin_amdgcn_cvt_pk_bf16_f32)
  v2bf r = __builtin_amdgcn_cvt_pk_bf16_f32(a, b);
  return __builtin_bit_cast(unsigned int, r);
#else
  return (unsigned int)f2bf(a) | ((unsigned int)f2bf(b) << 16);
#endif
}
__device__ __forceinline__ float bf2f(unsigned short s) {
  return __uint_as_float(((unsigned int)s) << 16);
}
__device__ __forceinline__ float bflo(unsigned int u) { return __uint_as_float(u << 16); }
__device__ __forceinline__ float bfhi(unsigned int u) { return __uint_as_float(u & 0xffff0000u); }

#define WMMA(ACC, A, Bv) \
  (ACC) = __builtin_amdgcn_wmma_f32_16x16x32_bf16(false, (A), false, (Bv), (short)0, (ACC), false, false)

// ---------------- workspace layout (bytes; all 64B aligned) ----------------
// W1  [672][512] bf16 : cols 0..639 = 10 experts x D1, 640..655 gate1_spe(all G), 656..665 gate1_sha, 666..671 pad
#define OFF_W1   0u
#define OFF_B1   688128u   // [672] f32
#define OFF_W2   690816u   // [320][64]  bf16 : row = e*32+f, k over D1
#define OFF_B2   731776u   // [320] f32
#define OFF_W4A  733056u   // [656][32]  bf16 : 0..511 spe2_W1(all), 512..639 sha2_W1, 640..655 gate2(all G)
#define OFF_B4A  775040u   // [656] f32
#define OFF_W4B  777664u   // [320][64]  bf16 : row = e*32+f, k over D1
#define OFF_B4B  818624u   // [320] f32
#define OFF_WT   819904u   // [256][32]  bf16 : row = d*64+f, k over D2
#define OFF_BT   836288u   // [256] f32
#define WS_BYTES 837312u

// ---------------- weight prep: fp32 -> bf16 transposed/concatenated ----------------
__global__ __launch_bounds__(256) void ple_prep(
    const float* __restrict__ spe1_W1, const float* __restrict__ spe1_b1,
    const float* __restrict__ spe1_W2, const float* __restrict__ spe1_b2,
    const float* __restrict__ sha1_W1, const float* __restrict__ sha1_b1,
    const float* __restrict__ sha1_W2, const float* __restrict__ sha1_b2,
    const float* __restrict__ g1sW, const float* __restrict__ g1sb,
    const float* __restrict__ g1hW, const float* __restrict__ g1hb,
    const float* __restrict__ spe2_W1, const float* __restrict__ spe2_b1,
    const float* __restrict__ spe2_W2, const float* __restrict__ spe2_b2,
    const float* __restrict__ sha2_W1, const float* __restrict__ sha2_b1,
    const float* __restrict__ sha2_W2, const float* __restrict__ sha2_b2,
    const float* __restrict__ g2sW, const float* __restrict__ g2sb,
    const float* __restrict__ tW1, const float* __restrict__ tb1,
    char* __restrict__ ws)
{
  int idx = blockIdx.x * 256 + threadIdx.x;

  unsigned short* W1  = (unsigned short*)(ws + OFF_W1);
  float*          B1  = (float*)(ws + OFF_B1);
  unsigned short* W2  = (unsigned short*)(ws + OFF_W2);
  float*          B2  = (float*)(ws + OFF_B2);
  unsigned short* W4A = (unsigned short*)(ws + OFF_W4A);
  float*          B4A = (float*)(ws + OFF_B4A);
  unsigned short* W4B = (unsigned short*)(ws + OFF_W4B);
  float*          B4B = (float*)(ws + OFF_B4B);
  unsigned short* WT  = (unsigned short*)(ws + OFF_WT);
  float*          BT  = (float*)(ws + OFF_BT);

  if (idx < 344064) {                       // W1 [672][512]
    int n = idx >> 9, k = idx & 511;
    float v = 0.f;
    if (n < 640) {
      int e = n >> 6, f = n & 63;           // e = s*4+g
      v = (e < 8) ? spe1_W1[((size_t)e * 512 + k) * 64 + f]
                  : sha1_W1[((size_t)(e - 8) * 512 + k) * 64 + f];
    } else if (n < 656) {
      int j = n - 640;                      // d*4 + kk
      v = g1sW[((size_t)(j >> 2) * 512 + k) * 4 + (j & 3)];
    } else if (n < 666) {
      v = g1hW[(size_t)k * 10 + (n - 656)];
    }
    W1[(size_t)n * 512 + k] = f2bf(v);
    return;
  }
  idx -= 344064;
  if (idx < 20480) {                        // W2 [320][64]
    int n = idx >> 6, k = idx & 63;
    int e = n >> 5, f = n & 31;
    float v = (e < 8) ? spe1_W2[((size_t)e * 64 + k) * 32 + f]
                      : sha1_W2[((size_t)(e - 8) * 64 + k) * 32 + f];
    W2[(size_t)n * 64 + k] = f2bf(v);
    return;
  }
  idx -= 20480;
  if (idx < 20992) {                        // W4A [656][32]
    int n = idx >> 5, k = idx & 31;
    float v = 0.f;
    if (n < 512) { int e = n >> 6, f = n & 63; v = spe2_W1[((size_t)e * 32 + k) * 64 + f]; }
    else if (n < 640) { int h = (n - 512) >> 6, f = (n - 512) & 63; v = sha2_W1[((size_t)h * 32 + k) * 64 + f]; }
    else { int j = n - 640; v = g2sW[((size_t)(j >> 2) * 32 + k) * 4 + (j & 3)]; }
    W4A[(size_t)n * 32 + k] = f2bf(v);
    return;
  }
  idx -= 20992;
  if (idx < 20480) {                        // W4B [320][64]
    int n = idx >> 6, k = idx & 63;
    int e = n >> 5, f = n & 31;
    float v = (e < 8) ? spe2_W2[((size_t)e * 64 + k) * 32 + f]
                      : sha2_W2[((size_t)(e - 8) * 64 + k) * 32 + f];
    W4B[(size_t)n * 64 + k] = f2bf(v);
    return;
  }
  idx -= 20480;
  if (idx < 8192) {                         // WT [256][32]
    int n = idx >> 5, k = idx & 31;
    int d = n >> 6, f = n & 63;
    WT[(size_t)n * 32 + k] = f2bf(tW1[((size_t)d * 32 + k) * 64 + f]);
    return;
  }
  idx -= 8192;
  if (idx < 672) {                          // B1
    int n = idx; float v = 0.f;
    if (n < 640) { int e = n >> 6, f = n & 63; v = (e < 8) ? spe1_b1[e * 64 + f] : sha1_b1[(e - 8) * 64 + f]; }
    else if (n < 656) v = g1sb[n - 640];
    else if (n < 666) v = g1hb[n - 656];
    B1[n] = v; return;
  }
  idx -= 672;
  if (idx < 320) {                          // B2
    int e = idx >> 5, f = idx & 31;
    B2[idx] = (e < 8) ? spe1_b2[e * 32 + f] : sha1_b2[(e - 8) * 32 + f];
    return;
  }
  idx -= 320;
  if (idx < 656) {                          // B4A
    int n = idx; float v;
    if (n < 512) { int e = n >> 6, f = n & 63; v = spe2_b1[e * 64 + f]; }
    else if (n < 640) { int h = (n - 512) >> 6, f = (n - 512) & 63; v = sha2_b1[h * 64 + f]; }
    else v = g2sb[n - 640];
    B4A[n] = v; return;
  }
  idx -= 656;
  if (idx < 320) {                          // B4B
    int e = idx >> 5, f = idx & 31;
    B4B[idx] = (e < 8) ? spe2_b2[e * 32 + f] : sha2_b2[(e - 8) * 32 + f];
    return;
  }
  idx -= 320;
  if (idx < 256) { BT[idx] = tb1[idx]; return; }   // tb1 [G,D1] flat = d*64+f
}

// ---------------- fused PLE kernel: 128 rows / block, 8 waves ----------------
// LDS (ushort units): sA   @0      128*520  x tile (bf16), later reused:
//                     sF   @0      128*336  feats / states (bf16)
//                     sI   @43008  128*80   spe|sha interest, later out (bf16)
//                     sH   @66560  128*680  H1 (bf16), later h2/ht (stride 664)
#define SMEM_BYTES 307200
__global__ __launch_bounds__(256) void ple_main(
    const float* __restrict__ x, const int* __restrict__ dom,
    const char* __restrict__ ws,
    const float* __restrict__ tW2, const float* __restrict__ tb2,
    float* __restrict__ out)
{
  extern __shared__ __align__(16) unsigned short smem[];
  unsigned short* sA = smem;
  unsigned short* sF = smem;
  unsigned short* sI = smem + 43008;
  unsigned short* sH = smem + 66560;

  const unsigned short* W1  = (const unsigned short*)(ws + OFF_W1);
  const float*          B1  = (const float*)(ws + OFF_B1);
  const unsigned short* W2  = (const unsigned short*)(ws + OFF_W2);
  const float*          B2  = (const float*)(ws + OFF_B2);
  const unsigned short* W4A = (const unsigned short*)(ws + OFF_W4A);
  const float*          B4A = (const float*)(ws + OFF_B4A);
  const unsigned short* W4B = (const unsigned short*)(ws + OFF_W4B);
  const float*          B4B = (const float*)(ws + OFF_B4B);
  const unsigned short* WT  = (const unsigned short*)(ws + OFF_WT);
  const float*          BT  = (const float*)(ws + OFF_BT);

  const int tid = threadIdx.x;
  const int w   = tid >> 5;            // wave 0..7 -> rows [16w,16w+16)
  const int l   = tid & 31;
  const int ln  = l & 15;              // lane-in-half: M (A) / N (B,C)
  const int lh  = l >> 4;              // half select
  const int akb = lh ? 8 : 0;          // A-frag K base
  const int bkb = lh ? 8 : 0;          // B-frag K base
  const int row0 = blockIdx.x * 128;
  const int wr0  = w << 4;
  const int mrow = wr0 + (lh ? 8 : 0); // C/D epilogue row base

  // ---- stage 0: x tile -> LDS bf16 (stride 520 to stagger banks) ----
  {
    const int r = tid >> 1, c0 = (tid & 1) << 8;
    const float* xr = x + (size_t)(row0 + r) * 512 + c0;
    unsigned short* dr = sA + r * 520 + c0;
#pragma unroll
    for (int c = 0; c < 256; c += 8) {
      f32x4 a = __builtin_nontemporal_load((const f32x4*)(xr + c));
      f32x4 b = __builtin_nontemporal_load((const f32x4*)(xr + c + 4));
      union { u16x8 u; unsigned int w32[4]; } o;
      o.w32[0] = f2bf_pk(a[0], a[1]);
      o.w32[1] = f2bf_pk(a[2], a[3]);
      o.w32[2] = f2bf_pk(b[0], b[1]);
      o.w32[3] = f2bf_pk(b[2], b[3]);
      *(u16x8*)(dr + c) = o.u;
    }
  }
  __syncthreads();

  // ---- stage 1: H1[128x672] = relu*(x @ W1cat + b1); gate logits kept linear ----
  {
    const unsigned short* arow = sA + (wr0 + ln) * 520 + akb;
    for (int nt = 0; nt < 42; ++nt) {
      const int n = nt * 16 + ln;
      const unsigned short* brow = W1 + (size_t)n * 512 + bkb;
      v8f acc = {};
#pragma unroll
      for (int kt = 0; kt < 16; ++kt) {
        frag16 a, b;
        a.p.lo = *(const u16x8*)(arow + kt * 32);
        a.p.hi = *(const u16x8*)(arow + kt * 32 + 16);
        b.p.lo = *(const u16x8*)(brow + kt * 32);
        b.p.hi = *(const u16x8*)(brow + kt * 32 + 16);
        WMMA(acc, a.h, b.h);
      }
      const float bias = B1[n];
      const bool rl = (n < 640);
      unsigned short* hrow = sH + mrow * 680 + n;
#pragma unroll
      for (int i = 0; i < 8; i += 2) {
        float v0 = acc[i] + bias, v1 = acc[i + 1] + bias;
        if (rl) { v0 = fmaxf(v0, 0.f); v1 = fmaxf(v1, 0.f); }
        const unsigned int p = f2bf_pk(v0, v1);
        hrow[i * 680] = (unsigned short)p;
        hrow[(i + 1) * 680] = (unsigned short)(p >> 16);
      }
    }
  }
  __syncthreads();

  // ---- stage 2: feats[128][10*32] = relu(H1_e @ W2_e + b2) per expert ----
  for (int e = 0; e < 10; ++e) {
    const unsigned short* arow = sH + (wr0 + ln) * 680 + e * 64 + akb;
#pragma unroll
    for (int half = 0; half < 2; ++half) {
      const int n = e * 32 + half * 16 + ln;
      const unsigned short* brow = W2 + (size_t)n * 64 + bkb;
      v8f acc = {};
#pragma unroll
      for (int kt = 0; kt < 2; ++kt) {
        frag16 a, b;
        a.p.lo = *(const u16x8*)(arow + kt * 32);
        a.p.hi = *(const u16x8*)(arow + kt * 32 + 16);
        b.p.lo = *(const u16x8*)(brow + kt * 32);
        b.p.hi = *(const u16x8*)(brow + kt * 32 + 16);
        WMMA(acc, a.h, b.h);
      }
      const float bias = B2[n];
      unsigned short* frow = sF + mrow * 336 + n;
#pragma unroll
      for (int i = 0; i < 8; i += 2) {
        const unsigned int p = f2bf_pk(fmaxf(acc[i] + bias, 0.f), fmaxf(acc[i + 1] + bias, 0.f));
        frow[i * 336] = (unsigned short)p;
        frow[(i + 1) * 336] = (unsigned short)(p >> 16);
      }
    }
  }
  __syncthreads();

  // ---- stage 3: per-row gating -> spe/sha interest (lane = row) ----
  if (tid < 128) {
    const int lr = tid;
    const int d = dom[row0 + lr];
    const unsigned short* hrow = sH + lr * 680;
    float g1[4], g1s[10];
    {
      const unsigned int* gp = (const unsigned int*)(hrow + 640 + 4 * d);
      const unsigned int u0 = gp[0], u1 = gp[1];
      g1[0] = bflo(u0); g1[1] = bfhi(u0); g1[2] = bflo(u1); g1[3] = bfhi(u1);
      float m = fmaxf(fmaxf(g1[0], g1[1]), fmaxf(g1[2], g1[3]));
      float s = 0.f;
#pragma unroll
      for (int k = 0; k < 4; ++k) { g1[k] = __expf(g1[k] - m); s += g1[k]; }
      s = 1.f / s;
#pragma unroll
      for (int k = 0; k < 4; ++k) g1[k] *= s;
    }
    {
      const unsigned int* gp = (const unsigned int*)(hrow + 656);
#pragma unroll
      for (int k = 0; k < 5; ++k) { const unsigned int u = gp[k]; g1s[2 * k] = bflo(u); g1s[2 * k + 1] = bfhi(u); }
      float m = -1e30f;
#pragma unroll
      for (int k = 0; k < 10; ++k) m = fmaxf(m, g1s[k]);
      float s = 0.f;
#pragma unroll
      for (int k = 0; k < 10; ++k) { g1s[k] = __expf(g1s[k] - m); s += g1s[k]; }
      s = 1.f / s;
#pragma unroll
      for (int k = 0; k < 10; ++k) g1s[k] *= s;
    }
    float si[32], sh[32];
#pragma unroll
    for (int f = 0; f < 32; ++f) { si[f] = 0.f; sh[f] = 0.f; }
    const unsigned short* frow = sF + lr * 336;
#pragma unroll
    for (int e = 0; e < 10; ++e) {
      const float gs = g1s[e];
      float gi = 0.f;
      if (e == d) gi = g1[0];
      else if (e == 4 + d) gi = g1[1];
      else if (e == 8) gi = g1[2];
      else if (e == 9) gi = g1[3];
      const unsigned int* f32p = (const unsigned int*)(frow + e * 32);
#pragma unroll
      for (int f = 0; f < 32; f += 2) {
        const unsigned int uv = f32p[f >> 1];
        const float v0 = bflo(uv), v1 = bfhi(uv);
        sh[f]     = fmaf(gs, v0, sh[f]);
        sh[f + 1] = fmaf(gs, v1, sh[f + 1]);
        si[f]     = fmaf(gi, v0, si[f]);
        si[f + 1] = fmaf(gi, v1, si[f + 1]);
      }
    }
    unsigned int* irow32 = (unsigned int*)(sI + lr * 80);
#pragma unroll
    for (int f = 0; f < 32; f += 2) {
      irow32[f >> 1]        = f2bf_pk(si[f], si[f + 1]);
      irow32[16 + (f >> 1)] = f2bf_pk(sh[f], sh[f + 1]);
    }
  }
  __syncthreads();

  // ---- stage 4: layer-2 hidden (all domains) + gate2 logits via WMMA ----
  {
    frag16 aspe, asha;
    const unsigned short* irow = sI + (wr0 + ln) * 80 + akb;
    aspe.p.lo = *(const u16x8*)(irow);
    aspe.p.hi = *(const u16x8*)(irow + 16);
    asha.p.lo = *(const u16x8*)(irow + 32);
    asha.p.hi = *(const u16x8*)(irow + 48);
    for (int nt = 0; nt < 41; ++nt) {
      const int n = nt * 16 + ln;
      const bool useSha = (nt >= 32) && (nt < 40);
      const unsigned short* brow = W4A + (size_t)n * 32 + bkb;
      frag16 b;
      b.p.lo = *(const u16x8*)(brow);
      b.p.hi = *(const u16x8*)(brow + 16);
      v8f acc = {};
      WMMA(acc, useSha ? asha.h : aspe.h, b.h);
      const float bias = B4A[n];
      const bool rl = (n < 640);
      unsigned short* hrow = sH + mrow * 664 + n;
#pragma unroll
      for (int i = 0; i < 8; i += 2) {
        float v0 = acc[i] + bias, v1 = acc[i + 1] + bias;
        if (rl) { v0 = fmaxf(v0, 0.f); v1 = fmaxf(v1, 0.f); }
        const unsigned int p = f2bf_pk(v0, v1);
        hrow[i * 664] = (unsigned short)p;
        hrow[(i + 1) * 664] = (unsigned short)(p >> 16);
      }
    }
  }

  // ---- stage 4b: layer-2 outputs (states) per expert via WMMA ----
  for (int e = 0; e < 10; ++e) {
    const unsigned short* arow = sH + (wr0 + ln) * 664 + e * 64 + akb;
#pragma unroll
    for (int half = 0; half < 2; ++half) {
      const int n = e * 32 + half * 16 + ln;
      const unsigned short* brow = W4B + (size_t)n * 64 + bkb;
      v8f acc = {};
#pragma unroll
      for (int kt = 0; kt < 2; ++kt) {
        frag16 a, b;
        a.p.lo = *(const u16x8*)(arow + kt * 32);
        a.p.hi = *(const u16x8*)(arow + kt * 32 + 16);
        b.p.lo = *(const u16x8*)(brow + kt * 32);
        b.p.hi = *(const u16x8*)(brow + kt * 32 + 16);
        WMMA(acc, a.h, b.h);
      }
      const float bias = B4B[n];
      unsigned short* frow = sF + mrow * 336 + n;
#pragma unroll
      for (int i = 0; i < 8; i += 2) {
        const unsigned int p = f2bf_pk(fmaxf(acc[i] + bias, 0.f), fmaxf(acc[i + 1] + bias, 0.f));
        frow[i * 336] = (unsigned short)p;
        frow[(i + 1) * 336] = (unsigned short)(p >> 16);
      }
    }
  }
  __syncthreads();

  // ---- stage 5: gate2 softmax + state mix -> out (lane = row) ----
  if (tid < 128) {
    const int lr = tid;
    const int d = dom[row0 + lr];
    const unsigned short* hrow = sH + lr * 664;
    float g2[4];
    {
      const unsigned int* gp = (const unsigned int*)(hrow + 640 + 4 * d);
      const unsigned int u0 = gp[0], u1 = gp[1];
      g2[0] = bflo(u0); g2[1] = bfhi(u0); g2[2] = bflo(u1); g2[3] = bfhi(u1);
    }
    float m = fmaxf(fmaxf(g2[0], g2[1]), fmaxf(g2[2], g2[3]));
    float s = 0.f;
#pragma unroll
    for (int k = 0; k < 4; ++k) { g2[k] = __expf(g2[k] - m); s += g2[k]; }
    s = 1.f / s;
    const int es[4] = { d, 4 + d, 8, 9 };
    float o[32];
#pragma unroll
    for (int f = 0; f < 32; ++f) o[f] = 0.f;
    const unsigned short* frow = sF + lr * 336;
#pragma unroll
    for (int k = 0; k < 4; ++k) {
      const float g = g2[k] * s;
      const unsigned int* f32p = (const unsigned int*)(frow + es[k] * 32);
#pragma unroll
      for (int f = 0; f < 32; f += 2) {
        const unsigned int uv = f32p[f >> 1];
        o[f]     = fmaf(g, bflo(uv), o[f]);
        o[f + 1] = fmaf(g, bfhi(uv), o[f + 1]);
      }
    }
    unsigned int* irow32 = (unsigned int*)(sI + lr * 80);
#pragma unroll
    for (int f = 0; f < 32; f += 2)
      irow32[f >> 1] = f2bf_pk(o[f], o[f + 1]);
  }
  __syncthreads();

  // ---- stage 6: tower layer-1 for all domains via WMMA ----
  {
    frag16 a;
    const unsigned short* irow = sI + (wr0 + ln) * 80 + akb;
    a.p.lo = *(const u16x8*)(irow);
    a.p.hi = *(const u16x8*)(irow + 16);
#pragma unroll
    for (int nt = 0; nt < 16; ++nt) {
      const int n = nt * 16 + ln;
      const unsigned short* brow = WT + (size_t)n * 32 + bkb;
      frag16 b;
      b.p.lo = *(const u16x8*)(brow);
      b.p.hi = *(const u16x8*)(brow + 16);
      v8f acc = {};
      WMMA(acc, a.h, b.h);
      const float bias = BT[n];
      unsigned short* hrow = sH + mrow * 664 + n;
#pragma unroll
      for (int i = 0; i < 8; i += 2) {
        const unsigned int p = f2bf_pk(fmaxf(acc[i] + bias, 0.f), fmaxf(acc[i + 1] + bias, 0.f));
        hrow[i * 664] = (unsigned short)p;
        hrow[(i + 1) * 664] = (unsigned short)(p >> 16);
      }
    }
  }
  __syncthreads();

  // ---- stage 7: tower layer-2 (64-dot, fp32 weights) ----
  if (tid < 128) {
    const int lr = tid;
    const int grow = row0 + lr;
    const int d = dom[grow];
    const unsigned int* h32 = (const unsigned int*)(sH + lr * 664 + d * 64);
    const float* tw = tW2 + d * 64;
    float y = tb2[d];
#pragma unroll
    for (int f = 0; f < 64; f += 2) {
      const unsigned int uv = h32[f >> 1];
      y = fmaf(bflo(uv), tw[f], y);
      y = fmaf(bfhi(uv), tw[f + 1], y);
    }
    __builtin_nontemporal_store(y, out + grow);
  }
}

// ---------------- launch ----------------
extern "C" void kernel_launch(void* const* d_in, const int* in_sizes, int n_in,
                              void* d_out, int out_size, void* d_ws, size_t ws_size,
                              hipStream_t stream) {
  (void)in_sizes; (void)n_in; (void)out_size; (void)ws_size;
  const float* x   = (const float*)d_in[0];
  const int*   dom = (const int*)d_in[1];
  const float* spe1_W1 = (const float*)d_in[2];
  const float* spe1_b1 = (const float*)d_in[3];
  const float* spe1_W2 = (const float*)d_in[4];
  const float* spe1_b2 = (const float*)d_in[5];
  const float* sha1_W1 = (const float*)d_in[6];
  const float* sha1_b1 = (const float*)d_in[7];
  const float* sha1_W2 = (const float*)d_in[8];
  const float* sha1_b2 = (const float*)d_in[9];
  const float* g1sW = (const float*)d_in[10];
  const float* g1sb = (const float*)d_in[11];
  const float* g1hW = (const float*)d_in[12];
  const float* g1hb = (const float*)d_in[13];
  const float* spe2_W1 = (const float*)d_in[14];
  const float* spe2_b1 = (const float*)d_in[15];
  const float* spe2_W2 = (const float*)d_in[16];
  const float* spe2_b2 = (const float*)d_in[17];
  const float* sha2_W1 = (const float*)d_in[18];
  const float* sha2_b1 = (const float*)d_in[19];
  const float* sha2_W2 = (const float*)d_in[20];
  const float* sha2_b2 = (const float*)d_in[21];
  const float* g2sW = (const float*)d_in[22];
  const float* g2sb = (const float*)d_in[23];
  const float* tW1 = (const float*)d_in[24];
  const float* tb1 = (const float*)d_in[25];
  const float* tW2 = (const float*)d_in[26];
  const float* tb2 = (const float*)d_in[27];
  char* ws = (char*)d_ws;

  // 416432 prep elements
  ple_prep<<<1627, 256, 0, stream>>>(
      spe1_W1, spe1_b1, spe1_W2, spe1_b2, sha1_W1, sha1_b1, sha1_W2, sha1_b2,
      g1sW, g1sb, g1hW, g1hb, spe2_W1, spe2_b1, spe2_W2, spe2_b2,
      sha2_W1, sha2_b1, sha2_W2, sha2_b2, g2sW, g2sb, tW1, tb1, ws);

  (void)hipFuncSetAttribute(reinterpret_cast<const void*>(ple_main),
                            hipFuncAttributeMaxDynamicSharedMemorySize, SMEM_BYTES);
  ple_main<<<32768 / 128, 256, SMEM_BYTES, stream>>>(x, dom, ws, tW2, tb2, (float*)d_out);
}
